// Informer_44006234915546
// MI455X (gfx1250) — compile-verified, matrix operand order checked by reading, not run on hardware
//
#include <hip/hip_runtime.h>
#include <hip/hip_bf16.h>

// ---------------------------------------------------------------------------
// Informer-style sparse attention for MI455X (gfx1250, wave32, WMMA).
// Dominant cost: streaming row-max of q@k^T (68.7 GFLOP) done with
// v_wmma_f32_16x16x32_bf16, fp32 accumulate.  Key tiles are shared across
// 8 waves per block through double-buffered LDS (8x cut in L2 traffic),
// staged with async global->LDS loads (ASYNCcnt) when available.
// ---------------------------------------------------------------------------

typedef __bf16 bf16_t;
typedef __attribute__((ext_vector_type(16))) __bf16 v16bf;
typedef __attribute__((ext_vector_type(8)))  float  v8f;
typedef int v4i __attribute__((vector_size(16)));   // matches async-LDS builtin param

#define BB   2
#define NN   8192
#define DD   256
#define UU   50
#define ROWS (BB * NN)          // 16384 flattened rows
#define NEG_BIG (-3.0e38f)

#define LDS_KSTRIDE 264         // padded row stride (bf16 elems) -> no bank conflicts
#define KSTEP 32                // keys staged per double-buffer slot

#define AS1 __attribute__((address_space(1)))
#define AS3 __attribute__((address_space(3)))

#if defined(__has_builtin)
#if __has_builtin(__builtin_amdgcn_global_load_async_to_lds_b128) && \
    __has_builtin(__builtin_amdgcn_s_wait_asynccnt)
#define USE_ASYNC_LDS 1
#endif
#endif

union ABOp { v16bf v; uint4 q[2]; };

// ---------------------------------------------------------------------------
// Kernel 1: fused QKV projection.  One wave (32 threads) computes one 16x16
// output tile of q/k/v = x @ W + b using 8x v_wmma_f32_16x16x32_bf16.
// Writes fp32 result and a bf16 copy (bf16 used by the WMMA score pass).
// grid = (ROWS/16, DD/16, 3), block = 32
// ---------------------------------------------------------------------------
__global__ __launch_bounds__(32) void proj_qkv(
    const float* __restrict__ x,
    const float* __restrict__ Wq, const float* __restrict__ bq,
    const float* __restrict__ Wk, const float* __restrict__ bk,
    const float* __restrict__ Wv, const float* __restrict__ bv,
    float* __restrict__ qf, float* __restrict__ kf, float* __restrict__ vf,
    bf16_t* __restrict__ qb, bf16_t* __restrict__ kb)
{
    const float* W;  const float* bias;  float* of;  bf16_t* ob;
    if (blockIdx.z == 0)      { W = Wq; bias = bq; of = qf; ob = qb;     }
    else if (blockIdx.z == 1) { W = Wk; bias = bk; of = kf; ob = kb;     }
    else                      { W = Wv; bias = bv; of = vf; ob = nullptr;}

    const int lane = threadIdx.x;
    const int nl   = lane & 15;      // column slot / A-row
    const int rsel = lane >> 4;      // which K half this lane holds
    const int row0 = blockIdx.x * 16;
    const int n0   = blockIdx.y * 16;

    // Stage W[:, n0:n0+16] panel into LDS as bf16, n-major, padded stride.
    __shared__ bf16_t wl[16 * LDS_KSTRIDE];
    __shared__ float  bl[16];
    for (int i = lane; i < 4096; i += 32) {
        int n = i & 15, k = i >> 4;
        wl[n * LDS_KSTRIDE + k] = (bf16_t)W[(size_t)k * DD + n0 + n];
    }
    if (lane < 16) bl[lane] = bias[n0 + lane];
    __syncthreads();

    v8f acc = {};
#pragma unroll
    for (int c = 0; c < 8; ++c) {
        // A operand: x row (row0+nl), K chunk c*32, per-lane K split
        // (lanes 0-15: K in {0..7,16..23}; lanes 16-31: K in {8..15,24..31}).
        const float* xp = x + (size_t)(row0 + nl) * DD + c * 32 + rsel * 8;
        float4 f0 = *(const float4*)(xp);
        float4 f1 = *(const float4*)(xp + 4);
        float4 f2 = *(const float4*)(xp + 16);
        float4 f3 = *(const float4*)(xp + 20);
        ABOp a;
        a.v[0]=(bf16_t)f0.x;  a.v[1]=(bf16_t)f0.y;  a.v[2]=(bf16_t)f0.z;  a.v[3]=(bf16_t)f0.w;
        a.v[4]=(bf16_t)f1.x;  a.v[5]=(bf16_t)f1.y;  a.v[6]=(bf16_t)f1.z;  a.v[7]=(bf16_t)f1.w;
        a.v[8]=(bf16_t)f2.x;  a.v[9]=(bf16_t)f2.y;  a.v[10]=(bf16_t)f2.z; a.v[11]=(bf16_t)f2.w;
        a.v[12]=(bf16_t)f3.x; a.v[13]=(bf16_t)f3.y; a.v[14]=(bf16_t)f3.z; a.v[15]=(bf16_t)f3.w;

        const bf16_t* wp = &wl[nl * LDS_KSTRIDE + c * 32 + rsel * 8];
        ABOp bm;
        bm.q[0] = *(const uint4*)(wp);
        bm.q[1] = *(const uint4*)(wp + 16);

        acc = __builtin_amdgcn_wmma_f32_16x16x32_bf16(
            false, a.v, false, bm.v, (short)0, acc, false, false);
    }

    const float bias_n = bl[nl];
#pragma unroll
    for (int j = 0; j < 8; ++j) {
        // C layout: vgpr j -> row M = j (lanes 0-15) or j+8 (lanes 16-31).
        int R = row0 + rsel * 8 + j;
        float v = acc[j] + bias_n;
        of[(size_t)R * DD + n0 + nl] = v;
        if (ob) ob[(size_t)R * DD + n0 + nl] = (bf16_t)v;
    }
}

// ---------------------------------------------------------------------------
// Kernel 2: ksum[b][d] = sum_m k[b][m][d]   (closed-form mean of score rows)
// grid = BB, block = DD
// ---------------------------------------------------------------------------
__global__ void ksum_kernel(const float* __restrict__ kf, float* __restrict__ ksum)
{
    int b = blockIdx.x, t = threadIdx.x;
    float s = 0.f;
    for (int m = 0; m < NN; ++m)
        s += kf[((size_t)b * NN + m) * DD + t];
    ksum[b * DD + t] = s;
}

// ---------------------------------------------------------------------------
// Kernel 3: streaming row-max of q@k^T via WMMA, then M = max - mean.
// 8 waves per block; each wave owns a 16-row query tile (128 rows/block).
// Key tiles (32 keys x 256 d, bf16) are staged into double-buffered LDS and
// shared by all 8 waves: 8x less L2 traffic than per-wave streaming.
// Next tile is staged asynchronously (ASYNCcnt) while WMMAs run.
// grid = ROWS/128, block = 256
// ---------------------------------------------------------------------------
__global__ __launch_bounds__(256) void rowmax_kernel(
    const bf16_t* __restrict__ qbf, const bf16_t* __restrict__ kbf,
    const float* __restrict__ qf,   const float* __restrict__ ksum,
    float* __restrict__ Mbuf)
{
    const int lane = threadIdx.x & 31;
    const int wv   = threadIdx.x >> 5;           // wave id 0..7
    const int nl   = lane & 15;
    const int rsel = lane >> 4;
    const int row0 = blockIdx.x * 128 + wv * 16; // this wave's query rows
    const int b    = blockIdx.x >> 6;            // 64 blocks per batch

    __shared__ bf16_t kl[2][KSTEP * LDS_KSTRIDE];  // 2 x 16.5KB key tiles
    __shared__ float  lmax[128];

    // Hold all 8 Q A-operands (full D=256) in registers: 64 VGPRs.
    ABOp aq[8];
#pragma unroll
    for (int c = 0; c < 8; ++c) {
        const bf16_t* p = qbf + (size_t)(row0 + nl) * DD + c * 32 + rsel * 8;
        aq[c].q[0] = *(const uint4*)(p);
        aq[c].q[1] = *(const uint4*)(p + 16);
    }

    const bf16_t* kbatch = kbf + (size_t)b * NN * DD;

    // Staging map: thread -> (row in tile, 32-elem column segment).
    const int sr = threadIdx.x >> 3;         // 0..31
    const int sc = (threadIdx.x & 7) * 32;   // 0,32,...,224

    // Prologue: stage tile 0 into buffer 0.
    {
        const bf16_t* gp = kbatch + (size_t)sr * DD + sc;
        bf16_t*       lp = &kl[0][sr * LDS_KSTRIDE + sc];
#if USE_ASYNC_LDS
#pragma unroll
        for (int i = 0; i < 4; ++i)
            __builtin_amdgcn_global_load_async_to_lds_b128(
                (AS1 v4i*)(gp + i * 8), (AS3 v4i*)(lp + i * 8), 0, 0);
        __builtin_amdgcn_s_wait_asynccnt(0);
#else
#pragma unroll
        for (int i = 0; i < 4; ++i)
            *(uint4*)(lp + i * 8) = *(const uint4*)(gp + i * 8);
#endif
    }
    __syncthreads();

    v8f runmax;
#pragma unroll
    for (int j = 0; j < 8; ++j) runmax[j] = NEG_BIG;

    for (int m0 = 0; m0 < NN; m0 += KSTEP) {
        const int cb = (m0 >> 5) & 1;        // current buffer
        const bool have_next = (m0 + KSTEP) < NN;
        const bf16_t* gp = kbatch + (size_t)(m0 + KSTEP + sr) * DD + sc;
        bf16_t*       lp = &kl[cb ^ 1][sr * LDS_KSTRIDE + sc];

#if USE_ASYNC_LDS
        if (have_next) {
#pragma unroll
            for (int i = 0; i < 4; ++i)
                __builtin_amdgcn_global_load_async_to_lds_b128(
                    (AS1 v4i*)(gp + i * 8), (AS3 v4i*)(lp + i * 8), 0, 0);
        }
#else
        uint4 st[4];
        if (have_next) {
#pragma unroll
            for (int i = 0; i < 4; ++i)
                st[i] = *(const uint4*)(gp + i * 8);   // overlap with WMMAs
        }
#endif

        // Compute on current buffer: two 16-key subtiles, 8 WMMAs each.
#pragma unroll
        for (int s2 = 0; s2 < 2; ++s2) {
            const bf16_t* kr = &kl[cb][(s2 * 16 + nl) * LDS_KSTRIDE];
            v8f acc = {};
#pragma unroll
            for (int c = 0; c < 8; ++c) {
                ABOp bm;
                bm.q[0] = *(const uint4*)(kr + c * 32 + rsel * 8);
                bm.q[1] = *(const uint4*)(kr + c * 32 + 16 + rsel * 8);
                acc = __builtin_amdgcn_wmma_f32_16x16x32_bf16(
                    false, aq[c].v, false, bm.v, (short)0, acc, false, false);
            }
#pragma unroll
            for (int j = 0; j < 8; ++j) runmax[j] = fmaxf(runmax[j], acc[j]);
        }

#if USE_ASYNC_LDS
        __builtin_amdgcn_s_wait_asynccnt(0);
#else
        if (have_next) {
#pragma unroll
            for (int i = 0; i < 4; ++i)
                *(uint4*)(lp + i * 8) = st[i];
        }
#endif
        __syncthreads();
    }

    // Reduce each row max across its 16-lane group (masks 1..8 stay in-group).
#pragma unroll
    for (int j = 0; j < 8; ++j) {
        float v = runmax[j];
        for (int m = 1; m < 16; m <<= 1)
            v = fmaxf(v, __shfl_xor(v, m, 32));
        runmax[j] = v;
    }

    if (lane == 0)
        for (int j = 0; j < 8; ++j) lmax[wv * 16 + j] = runmax[j];       // rows 0..7
    if (lane == 16)
        for (int j = 0; j < 8; ++j) lmax[wv * 16 + 8 + j] = runmax[j];   // rows 8..15
    __syncthreads();

    if (lane < 16) {
        int R = row0 + lane;
        const float* qr = qf + (size_t)R * DD;
        const float* ks = ksum + b * DD;
        float dot = 0.f;
        for (int d = 0; d < DD; ++d) dot += qr[d] * ks[d];
        const float inv_s = 0.17677669529663687f;              // 1/sqrt(32)
        Mbuf[R] = lmax[wv * 16 + lane] * inv_s - (dot * inv_s) * (1.0f / NN);
    }
}

// ---------------------------------------------------------------------------
// Kernel 4: top-u (u=50) indices per batch by iterative argmax over M.
// grid = BB, block = 256.  (Mbuf is recomputed every call, so in-place
// invalidation is deterministic across graph replays.)
// ---------------------------------------------------------------------------
__global__ __launch_bounds__(256) void topk_kernel(float* __restrict__ Mbuf,
                                                   int* __restrict__ idxbuf)
{
    int b = blockIdx.x, t = threadIdx.x;
    __shared__ float sv[256];
    __shared__ int   si[256];
    float* Mb = Mbuf + (size_t)b * NN;
    for (int it = 0; it < UU; ++it) {
        float best = NEG_BIG; int bi = 0;
        for (int m = t; m < NN; m += 256) {
            float v = Mb[m];
            if (v > best) { best = v; bi = m; }
        }
        sv[t] = best; si[t] = bi;
        __syncthreads();
        for (int s = 128; s > 0; s >>= 1) {
            if (t < s && sv[t + s] > sv[t]) { sv[t] = sv[t + s]; si[t] = si[t + s]; }
            __syncthreads();
        }
        if (t == 0) { idxbuf[b * 64 + it] = si[0]; Mb[si[0]] = NEG_BIG; }
        __syncthreads();
    }
}

// ---------------------------------------------------------------------------
// Kernel 5: zero the pooled accumulator (workspace is not re-poisoned).
// ---------------------------------------------------------------------------
__global__ void init_pooled(float* __restrict__ pooled)
{
    int i = threadIdx.x;
    if (i < BB * DD) pooled[i] = 0.f;
}

// ---------------------------------------------------------------------------
// Kernel 6: second attention pass for the u selected queries.
// One block per (b, j): logits over all 8192 keys kept in 32KB LDS,
// softmax (no scaling, per reference), context = attn @ v, and the 1/u
// mean folded into an fp32 atomic accumulate into pooled[b][:].
// grid = BB*UU, block = 256
// ---------------------------------------------------------------------------
__global__ __launch_bounds__(256) void attn2_kernel(
    const float* __restrict__ qf, const float* __restrict__ kf,
    const float* __restrict__ vf, const int* __restrict__ idxbuf,
    float* __restrict__ pooled)
{
    int b = blockIdx.x / UU;
    int j = blockIdx.x % UU;
    int t = threadIdx.x;

    __shared__ float qs[DD];
    __shared__ float red[256];
    __shared__ float lg[NN];

    int qi = idxbuf[b * 64 + j];
    qs[t] = qf[((size_t)b * NN + qi) * DD + t];
    __syncthreads();

    // logits + thread-local max
    float lmaxv = NEG_BIG;
    for (int m = t; m < NN; m += 256) {
        const float* kr = kf + ((size_t)b * NN + m) * DD;
        float dot = 0.f;
        for (int d = 0; d < DD; ++d) dot += qs[d] * kr[d];
        lg[m] = dot;
        lmaxv = fmaxf(lmaxv, dot);
    }
    red[t] = lmaxv;
    __syncthreads();
    for (int s = 128; s > 0; s >>= 1) {
        if (t < s) red[t] = fmaxf(red[t], red[t + s]);
        __syncthreads();
    }
    float mx = red[0];
    __syncthreads();

    // exp weights in place + sum
    float lsum = 0.f;
    for (int m = t; m < NN; m += 256) {
        float w = __expf(lg[m] - mx);
        lg[m] = w;
        lsum += w;
    }
    red[t] = lsum;
    __syncthreads();
    for (int s = 128; s > 0; s >>= 1) {
        if (t < s) red[t] += red[t + s];
        __syncthreads();
    }
    float Z = red[0];
    __syncthreads();

    // context[d=t] = sum_m w_m * v[m][d]   (v reads coalesced across threads)
    float ctx = 0.f;
    for (int m = 0; m < NN; ++m)
        ctx += lg[m] * vf[((size_t)b * NN + m) * DD + t];

    atomicAdd(&pooled[b * DD + t], ctx * (1.0f / Z) * (1.0f / UU));
}

// ---------------------------------------------------------------------------
// Kernel 7: out = pooled @ Wp + bp.  grid = BB, block = 256.
// ---------------------------------------------------------------------------
__global__ __launch_bounds__(256) void outproj_kernel(
    const float* __restrict__ pooled, const float* __restrict__ Wp,
    const float* __restrict__ bp, float* __restrict__ out)
{
    int b = blockIdx.x, t = threadIdx.x;
    __shared__ float ps[DD];
    ps[t] = pooled[b * DD + t];
    __syncthreads();
    float acc = bp[t];
    for (int d = 0; d < DD; ++d)
        acc += ps[d] * Wp[(size_t)d * DD + t];
    out[b * DD + t] = acc;
}

// ---------------------------------------------------------------------------
extern "C" void kernel_launch(void* const* d_in, const int* in_sizes, int n_in,
                              void* d_out, int out_size, void* d_ws, size_t ws_size,
                              hipStream_t stream)
{
    const float* x  = (const float*)d_in[0];
    const float* Wq = (const float*)d_in[1];
    const float* bq = (const float*)d_in[2];
    const float* Wk = (const float*)d_in[3];
    const float* bk = (const float*)d_in[4];
    const float* Wv = (const float*)d_in[5];
    const float* bv = (const float*)d_in[6];
    const float* Wp = (const float*)d_in[7];
    const float* bp = (const float*)d_in[8];
    float* out = (float*)d_out;

    char* ws = (char*)d_ws;
    size_t off = 0;
    auto carve = [&](size_t bytes) -> void* {
        void* p = ws + off;
        off = (off + bytes + 255) & ~(size_t)255;
        return p;
    };

    float*  qf     = (float*) carve((size_t)ROWS * DD * sizeof(float));   // 16 MB
    float*  kf     = (float*) carve((size_t)ROWS * DD * sizeof(float));   // 16 MB
    float*  vf     = (float*) carve((size_t)ROWS * DD * sizeof(float));   // 16 MB
    bf16_t* qb     = (bf16_t*)carve((size_t)ROWS * DD * sizeof(bf16_t));  //  8 MB
    bf16_t* kb     = (bf16_t*)carve((size_t)ROWS * DD * sizeof(bf16_t));  //  8 MB
    float*  ksum   = (float*) carve((size_t)BB * DD * sizeof(float));
    float*  Mbuf   = (float*) carve((size_t)ROWS * sizeof(float));
    int*    idxbuf = (int*)   carve((size_t)BB * 64 * sizeof(int));
    float*  pooled = (float*) carve((size_t)BB * DD * sizeof(float));
    (void)ws_size; (void)in_sizes; (void)n_in; (void)out_size;

    // 1) fused QKV projection (WMMA bf16)
    dim3 gproj(ROWS / 16, DD / 16, 3);
    proj_qkv<<<gproj, 32, 0, stream>>>(x, Wq, bq, Wk, bk, Wv, bv,
                                       qf, kf, vf, qb, kb);
    // 2) key column sums (closed-form mean of score rows)
    ksum_kernel<<<BB, DD, 0, stream>>>(kf, ksum);
    // 3) streaming row-max of q@k^T (the heavy WMMA pass), M = max - mean
    rowmax_kernel<<<ROWS / 128, 256, 0, stream>>>(qb, kb, qf, ksum, Mbuf);
    // 4) top-u indices per batch
    topk_kernel<<<BB, 256, 0, stream>>>(Mbuf, idxbuf);
    // 5) zero pooled accumulator
    init_pooled<<<1, 512, 0, stream>>>(pooled);
    // 6) second attention pass on reduced queries, mean-pooled via atomics
    attn2_kernel<<<BB * UU, 256, 0, stream>>>(qf, kf, vf, idxbuf, pooled);
    // 7) output projection
    outproj_kernel<<<BB, 256, 0, stream>>>(pooled, Wp, bp, out);
}
